// QuantCSNN_44495861187161
// MI455X (gfx1250) — matrix operand order repaired and checked by reading, not compiled, not run on hardware
//
#include <hip/hip_runtime.h>
#include <stdint.h>

typedef __attribute__((ext_vector_type(16))) _Float16 v16h;
typedef __attribute__((ext_vector_type(8)))  float    v8f;

#define T_STEPS 8
#define BATCH   128
#define CH      128
#define IMH     28
#define IMW     28
#define PH      14
#define PW      14
#define K_FC    (CH*PH*PW)     // 25088
#define N_OUT   10
#define KT      (K_FC/32)      // 784 K-tiles of 32
#define ROWS    (T_STEPS*BATCH) // 1024 GEMM rows

// ---------------------------------------------------------------------------
// Kernel 1: conv3x3(pad1, 1 in-ch) + BN(eval) + IF-node-1 over T=8 (constant
// input -> closed-form per-pixel recurrence) + 2x2 maxpool(OR) -> bit-packed
// spikes sp[b][k], bit t = pooled spike at timestep t.  k = c*196 + ph*14 + pw
// ---------------------------------------------------------------------------
__global__ void k_conv_if_pool(const float* __restrict__ x,
                               const float* __restrict__ cw,
                               const float* __restrict__ gamma,
                               const float* __restrict__ beta,
                               const float* __restrict__ mean,
                               const float* __restrict__ var,
                               uint8_t* __restrict__ sp)
{
    int idx = blockIdx.x * blockDim.x + threadIdx.x;   // = b*K_FC + k
    if (idx >= BATCH * K_FC) return;
    int b  = idx / K_FC;
    int k  = idx - b * K_FC;
    int c  = k / (PH * PW);
    int p  = k - c * (PH * PW);
    int ph = p / PW, pw = p - ph * PW;

    float w[9];
#pragma unroll
    for (int i = 0; i < 9; ++i) w[i] = cw[c * 9 + i];
    float sc = gamma[c] * rsqrtf(var[c] + 1e-5f);
    float mu = mean[c], bt = beta[c];

    const float* xb = x + (size_t)b * IMH * IMW;
    float xs[4][4];
#pragma unroll
    for (int dy = 0; dy < 4; ++dy) {
        int h = 2 * ph - 1 + dy;
#pragma unroll
        for (int dx = 0; dx < 4; ++dx) {
            int wc = 2 * pw - 1 + dx;
            xs[dy][dx] = (h >= 0 && h < IMH && wc >= 0 && wc < IMW)
                         ? xb[h * IMW + wc] : 0.0f;
        }
    }

    float pre[4];
#pragma unroll
    for (int oy = 0; oy < 2; ++oy)
#pragma unroll
        for (int ox = 0; ox < 2; ++ox) {
            float acc = 0.0f;
#pragma unroll
            for (int ky = 0; ky < 3; ++ky)
#pragma unroll
                for (int kx = 0; kx < 3; ++kx)
                    acc += xs[oy + ky][ox + kx] * w[ky * 3 + kx];
            pre[oy * 2 + ox] = (acc - mu) * sc + bt;
        }

    uint32_t bits = 0;
    float v0 = 0.f, v1 = 0.f, v2 = 0.f, v3 = 0.f;
#pragma unroll
    for (int t = 0; t < T_STEPS; ++t) {
        v0 += pre[0]; v1 += pre[1]; v2 += pre[2]; v3 += pre[3];
        int s0 = v0 >= 1.0f, s1 = v1 >= 1.0f, s2 = v2 >= 1.0f, s3 = v3 >= 1.0f;
        if (s0) v0 = 0.f;
        if (s1) v1 = 0.f;
        if (s2) v2 = 0.f;
        if (s3) v3 = 0.f;
        bits |= ((uint32_t)(s0 | s1 | s2 | s3)) << t;
    }
    sp[idx] = (uint8_t)bits;
}

// ---------------------------------------------------------------------------
// Kernel 2: fc_w (f32 [10][25088]) -> f16, pre-swizzled into the per-lane
// WMMA B-matrix (32x16) layout: wh[kt][lane][j], lane: N = lane&15,
// K = kt*32 + ((j&8)<<1) + ((lane&16)>>1) + (j&7).  One contiguous 32B vector load per lane/tile.
// ---------------------------------------------------------------------------
__global__ void k_pack_w(const float* __restrict__ fcw, _Float16* __restrict__ wh)
{
    int idx = blockIdx.x * blockDim.x + threadIdx.x;
    if (idx >= KT * 32 * 16) return;
    int kt = idx >> 9;          // / 512
    int l  = (idx >> 4) & 31;
    int j  = idx & 15;
    int n  = l & 15;
    int k  = kt * 32 + ((j & 8) << 1) + ((l & 16) >> 1) + (j & 7);
    float v = (n < N_OUT) ? fcw[(size_t)n * K_FC + k] : 0.0f;
    wh[idx] = (_Float16)v;
}

// ---------------------------------------------------------------------------
// Kernel 3: batched FC GEMM  Z[t*128+b][n] = sum_k f[t][b][k] * W[n][k]
// M = 1024 rows (one wave per 16-row tile, t is wave-uniform), N = 16, K = 25088.
// A built on the fly from packed spike bits per the documented 16-bit A layout:
//   lanes 0-15 : elems 0-7 -> K 0-7,  elems 8-15 -> K 16-23
//   lanes 16-31: elems 0-7 -> K 8-15, elems 8-15 -> K 24-31
// ---------------------------------------------------------------------------
__global__ void k_fc_wmma(const uint8_t* __restrict__ sp,
                          const _Float16* __restrict__ wh,
                          float* __restrict__ z)
{
    int lane = threadIdx.x & 31;
    int wave = (blockIdx.x * blockDim.x + threadIdx.x) >> 5;  // 0..63
    int r0   = wave * 16;
    int t    = r0 >> 7;                 // timestep (wave-uniform)
    int m    = lane & 15;
    int b    = (r0 & 127) + m;          // batch row for this lane's A data
    int koffA = (lane & 16) ? 8 : 0;

    const uint8_t* sprow = sp + (size_t)b * K_FC;
    v8f acc = {};

    for (int kt = 0; kt < KT; ++kt) {
        int kb = kt * 32;
        uint64_t lo = *(const uint64_t*)(sprow + kb + koffA);        // K kb+koffA .. +7
        uint64_t hi = *(const uint64_t*)(sprow + kb + 16 + koffA);   // K kb+16+koffA .. +7
        lo >>= t;
        hi >>= t;
        v16h a;
#pragma unroll
        for (int j = 0; j < 8; ++j) {
            a[j]     = (_Float16)(int)((lo >> (8 * j)) & 1u);
            a[j + 8] = (_Float16)(int)((hi >> (8 * j)) & 1u);
        }
        v16h bm = *(const v16h*)(wh + (size_t)kt * 512 + lane * 16);

        __builtin_prefetch(wh + (size_t)(kt + 1) * 512 + lane * 16, 0, 1);
        __builtin_prefetch(sprow + kb + 32, 0, 1);

        acc = __builtin_amdgcn_wmma_f32_16x16x32_f16(
            /*neg_a=*/false, a, /*neg_b=*/false, bm,
            /*c_mod=*/(short)0, acc, /*reuse_a=*/false, /*reuse_b=*/false);
    }

    // C/D layout: lanes 0-15: N=lane, M=g; lanes 16-31: N=lane-16, M=8+g
    int n   = lane & 15;
    int mhi = (lane & 16) ? 8 : 0;
#pragma unroll
    for (int g = 0; g < 8; ++g)
        z[(size_t)(r0 + mhi + g) * 16 + n] = acc[g];
}

// ---------------------------------------------------------------------------
// Kernel 4: IF-node-2 recurrence over T on [128,10] + firing-rate mean
// ---------------------------------------------------------------------------
__global__ void k_if2(const float* __restrict__ z, float* __restrict__ out)
{
    int idx = blockIdx.x * blockDim.x + threadIdx.x;
    if (idx >= BATCH * N_OUT) return;
    int b = idx / N_OUT;
    int n = idx - b * N_OUT;
    float v = 0.0f, acc = 0.0f;
#pragma unroll
    for (int t = 0; t < T_STEPS; ++t) {
        v += z[(size_t)(t * BATCH + b) * 16 + n];
        if (v >= 1.0f) { acc += 1.0f; v = 0.0f; }
    }
    out[idx] = acc * (1.0f / T_STEPS);
}

// ---------------------------------------------------------------------------
extern "C" void kernel_launch(void* const* d_in, const int* in_sizes, int n_in,
                              void* d_out, int out_size, void* d_ws, size_t ws_size,
                              hipStream_t stream)
{
    const float* x     = (const float*)d_in[0];
    const float* cw    = (const float*)d_in[1];
    const float* gamma = (const float*)d_in[2];
    const float* beta  = (const float*)d_in[3];
    const float* mean  = (const float*)d_in[4];
    const float* var   = (const float*)d_in[5];
    const float* fcw   = (const float*)d_in[6];
    float* out = (float*)d_out;

    // workspace layout (total ~4.08 MB, all 32B aligned)
    uint8_t*  ws = (uint8_t*)d_ws;
    uint8_t*  sp = ws;                                         // 3,211,264 B
    _Float16* wh = (_Float16*)(ws + (size_t)BATCH * K_FC);     //   802,816 B
    float*    z  = (float*)(ws + (size_t)BATCH * K_FC
                               + (size_t)KT * 32 * 16 * sizeof(_Float16)); // 65,536 B

    k_conv_if_pool<<<(BATCH * K_FC + 255) / 256, 256, 0, stream>>>(
        x, cw, gamma, beta, mean, var, sp);

    k_pack_w<<<(KT * 32 * 16 + 255) / 256, 256, 0, stream>>>(fcw, wh);

    // 64 waves: 16 blocks x 128 threads (4 waves/block), full waves, no divergence
    k_fc_wmma<<<16, 128, 0, stream>>>(sp, wh, z);

    k_if2<<<(BATCH * N_OUT + 255) / 256, 256, 0, stream>>>(z, out);
}
